// SelectSSM_15247133901505
// MI455X (gfx1250) — matrix operand order repaired and checked by reading, not compiled, not run on hardware
//
#include <hip/hip_runtime.h>

// ---------------------------------------------------------------------------
// Types for CDNA5 WMMA (wave32)
// ---------------------------------------------------------------------------
typedef __bf16 bf16_t;
typedef __attribute__((ext_vector_type(16))) __bf16 v16bf;
typedef __attribute__((ext_vector_type(8)))  __bf16 v8bf;
typedef __attribute__((ext_vector_type(8)))  float  v8f;

union BF16x16 { v16bf v; v8bf h[2]; };

// Problem constants
#define BB      4
#define LL      4096
#define DM      1024
#define DI      1024
#define MROWS   (BB * LL)      // 16384
#define NCHUNK  32
#define CHLEN   (LL / NCHUNK)  // 128

// ---------------------------------------------------------------------------
// A-fragment (16x32 bf16, row-major, lda=K):
//   lane<16: K = k0+0..7 and k0+16..23 ; lane>=16: K = k0+8..15 and k0+24..31
// Each half is 8 bf16 = 16 bytes -> one b128 load.
// ---------------------------------------------------------------------------
__device__ inline v16bf load_a_frag(const bf16_t* __restrict__ A, int K, int row,
                                    int k0, int lane) {
  int kb = k0 + ((lane & 16) ? 8 : 0);
  const bf16_t* p = A + (size_t)row * K + kb;
  BF16x16 u;
  u.h[0] = *(const v8bf*)(p);
  u.h[1] = *(const v8bf*)(p + 16);
  return u.v;
}

// B-fragment (32x16 bf16; B stored row-major as (N,K)):
//   lane<16: col=n0+lane, K = k0+0..7 and k0+16..23 ; lane>=16: col=n0+lane-16, K = k0+8..15 and k0+24..31
// two 16-byte runs at +0 and +16 elements -> two b128 loads.
__device__ inline v16bf load_b_frag(const bf16_t* __restrict__ Bm, int K, int col,
                                    int k0, int lane) {
  int kb = k0 + ((lane & 16) ? 8 : 0);
  const bf16_t* p = Bm + (size_t)col * K + kb;
  BF16x16 u;
  u.h[0] = *(const v8bf*)(p);
  u.h[1] = *(const v8bf*)(p + 16);
  return u.v;
}

// ---------------------------------------------------------------------------
// C[m,n] = sum_k A[m,k] * B[n,k]   (A: MxK bf16, B: NxK bf16, C: MxN f32)
// One wave -> 32 x (16*BN) output tile; M mult of 32, N mult of 16*BN,
// K mult of 32.  BN=4 -> 8 wmma : 12 b128 loads per K-step.
// ---------------------------------------------------------------------------
template <int BN>
__global__ void gemm_bf16_bt(const bf16_t* __restrict__ A,
                             const bf16_t* __restrict__ Bm,
                             float* __restrict__ C, int M, int N, int K) {
  int lane = threadIdx.x & 31;
  int wave = blockIdx.x * (blockDim.x >> 5) + (threadIdx.x >> 5);
  int tiles_n = N / (16 * BN);
  int total   = (M >> 5) * tiles_n;
  if (wave >= total) return;  // wave-uniform: EXEC stays all-ones for WMMA
  int tm = wave / tiles_n;
  int tn = wave - tm * tiles_n;
  int m0 = tm << 5, n0 = tn * (16 * BN);

  v8f acc[2][BN];
#pragma unroll
  for (int i = 0; i < 2; ++i)
#pragma unroll
    for (int j = 0; j < BN; ++j) acc[i][j] = (v8f){};

  int rA0 = m0 + (lane & 15);
  int rA1 = rA0 + 16;
  int cB  = n0 + (lane & 15);

  for (int k0 = 0; k0 < K; k0 += 32) {
    // keep L2 ahead on the streaming A operand (global_prefetch_b8)
    if (k0 + 128 < K) {
      __builtin_prefetch(A + (size_t)rA0 * K + k0 + 128, 0, 1);
      __builtin_prefetch(A + (size_t)rA1 * K + k0 + 128, 0, 1);
    }
    v16bf a0 = load_a_frag(A, K, rA0, k0, lane);
    v16bf a1 = load_a_frag(A, K, rA1, k0, lane);
    v16bf b[BN];
#pragma unroll
    for (int j = 0; j < BN; ++j)
      b[j] = load_b_frag(Bm, K, cB + 16 * j, k0, lane);
#pragma unroll
    for (int j = 0; j < BN; ++j) {
      acc[0][j] = __builtin_amdgcn_wmma_f32_16x16x32_bf16(
          false, a0, false, b[j], (short)0, acc[0][j], false, false);
      acc[1][j] = __builtin_amdgcn_wmma_f32_16x16x32_bf16(
          false, a1, false, b[j], (short)0, acc[1][j], false, false);
    }
  }

  // C/D layout: VGPR v holds (M = v + (lane<16?0:8), N = lane&15)
  int rb = (lane & 16) ? 8 : 0;
  int cc = lane & 15;
#pragma unroll
  for (int i = 0; i < 2; ++i)
#pragma unroll
    for (int j = 0; j < BN; ++j)
#pragma unroll
      for (int v = 0; v < 8; ++v)
        C[(size_t)(m0 + 16 * i + rb + v) * N + (n0 + 16 * j + cc)] = acc[i][j][v];
}

// ---------------------------------------------------------------------------
// Elementwise converts
// ---------------------------------------------------------------------------
__global__ void f32_to_bf16_k(const float* __restrict__ in, bf16_t* __restrict__ out,
                              int n) {
  int i = blockIdx.x * blockDim.x + threadIdx.x;
  if (i < n) out[i] = (bf16_t)in[i];
}

// zero-pad x_proj_weight (66x1024) -> (96x1024) bf16
__global__ void pad_xproj_k(const float* __restrict__ in, bf16_t* __restrict__ out,
                            int rows_in, int rows_out, int K) {
  int i = blockIdx.x * blockDim.x + threadIdx.x;
  int n = rows_out * K;
  if (i >= n) return;
  int r = i / K, k = i - r * K;
  out[i] = (bf16_t)((r < rows_in) ? in[r * K + k] : 0.0f);
}

// take first `cols` of each xdbl row (ld=96) -> packed bf16 (for dt GEMM)
__global__ void slice_bf16_k(const float* __restrict__ xdbl, bf16_t* __restrict__ out,
                             int M, int ld, int cols) {
  int i = blockIdx.x * blockDim.x + threadIdx.x;
  if (i >= M * cols) return;
  int m = i / cols, c = i - m * cols;
  out[i] = (bf16_t)xdbl[(size_t)m * ld + c];
}

// ---------------------------------------------------------------------------
// Causal 2-tap depthwise conv + bias + SiLU (reads xc half of xz)
// out[b,l,d] = silu(w0*x[l-1] + w1*x[l] + b_d); writes f32 and bf16 copies.
// ---------------------------------------------------------------------------
__global__ void conv_silu_k(const float* __restrict__ xz,
                            const float* __restrict__ cw,
                            const float* __restrict__ cb,
                            float* __restrict__ xc_f32,
                            bf16_t* __restrict__ xc_bf) {
  int i = blockIdx.x * blockDim.x + threadIdx.x;
  if (i >= MROWS * DI) return;
  int d = i & (DI - 1);
  int m = i >> 10;          // b*L + l
  int l = m & (LL - 1);
  float cur  = xz[(size_t)m * (2 * DI) + d];
  float prev = (l > 0) ? xz[(size_t)(m - 1) * (2 * DI) + d] : 0.0f;
  float v = cw[d * 2 + 0] * prev + cw[d * 2 + 1] * cur + cb[d];
  v = v / (1.0f + expf(-v));          // SiLU
  xc_f32[i] = v;
  xc_bf[i]  = (bf16_t)v;
}

// ---------------------------------------------------------------------------
// Chunked selective scan (D_STATE==1).  h' = exp(delta*A)*h + delta*Bs*u
// pass1: per (b,d,chunk): product of a's and chunk-local h (h_in=0)
// pass2: per (b,d): serial prefix over 32 chunks -> h_in per chunk
// pass3: replay with correct h_in, emit y = h*Cs + u*Ds
// ---------------------------------------------------------------------------
__device__ inline float softplus_f(float x) {
  return (x > 20.0f) ? x : log1pf(expf(x));
}

__global__ void scan_pass1_k(const float* __restrict__ dts_full,
                             const float* __restrict__ xdbl,
                             const float* __restrict__ xcc,
                             const float* __restrict__ dtb,
                             const float* __restrict__ Alogs,
                             float* __restrict__ chP,
                             float* __restrict__ chH) {
  int i = blockIdx.x * blockDim.x + threadIdx.x;   // ((b*NCHUNK)+c)*DI + d
  if (i >= BB * NCHUNK * DI) return;
  int d = i & (DI - 1);
  int t = i >> 10;
  int c = t & (NCHUNK - 1);
  int b = t >> 5;
  float Ad   = -expf(Alogs[d]);
  float bias = dtb[d];
  float h = 0.0f, P = 1.0f;
  int l0 = c * CHLEN;
  for (int j = 0; j < CHLEN; ++j) {
    size_t m  = (size_t)b * LL + (l0 + j);
    float dt  = softplus_f(dts_full[m * DI + d] + bias);
    float a   = expf(dt * Ad);
    float bu  = dt * xdbl[m * 96 + 64] * xcc[m * DI + d];
    h = a * h + bu;
    P *= a;
  }
  chP[(size_t)(b * DI + d) * NCHUNK + c] = P;
  chH[(size_t)(b * DI + d) * NCHUNK + c] = h;
}

__global__ void scan_pass2_k(const float* __restrict__ chP,
                             const float* __restrict__ chH,
                             float* __restrict__ hin) {
  int i = blockIdx.x * blockDim.x + threadIdx.x;   // b*DI + d
  if (i >= BB * DI) return;
  float h = 0.0f;
  for (int c = 0; c < NCHUNK; ++c) {
    hin[(size_t)i * NCHUNK + c] = h;
    h = chP[(size_t)i * NCHUNK + c] * h + chH[(size_t)i * NCHUNK + c];
  }
}

__global__ void scan_pass3_k(const float* __restrict__ dts_full,
                             const float* __restrict__ xdbl,
                             const float* __restrict__ xcc,
                             const float* __restrict__ dtb,
                             const float* __restrict__ Alogs,
                             const float* __restrict__ Ds,
                             const float* __restrict__ hin,
                             float* __restrict__ y) {
  int i = blockIdx.x * blockDim.x + threadIdx.x;
  if (i >= BB * NCHUNK * DI) return;
  int d = i & (DI - 1);
  int t = i >> 10;
  int c = t & (NCHUNK - 1);
  int b = t >> 5;
  float Ad   = -expf(Alogs[d]);
  float bias = dtb[d];
  float Dd   = Ds[d];
  float h = hin[(size_t)(b * DI + d) * NCHUNK + c];
  int l0 = c * CHLEN;
  for (int j = 0; j < CHLEN; ++j) {
    size_t m  = (size_t)b * LL + (l0 + j);
    float u   = xcc[m * DI + d];
    float dt  = softplus_f(dts_full[m * DI + d] + bias);
    float a   = expf(dt * Ad);
    float bu  = dt * xdbl[m * 96 + 64] * u;
    h = a * h + bu;
    float Cs = xdbl[m * 96 + 65];
    y[m * DI + d] = h * Cs + u * Dd;
  }
}

// ---------------------------------------------------------------------------
// LayerNorm over D_INNER + silu(z) gate; writes bf16 operand for final GEMM.
// One 256-thread block per (b,l) row.
// ---------------------------------------------------------------------------
__global__ void ln_gate_k(const float* __restrict__ y,
                          const float* __restrict__ xz,
                          const float* __restrict__ lnw,
                          const float* __restrict__ lnb,
                          bf16_t* __restrict__ yln) {
  int row = blockIdx.x;            // b*L + l
  int t   = threadIdx.x;           // 0..255
  const float* yr = y + (size_t)row * DI;
  float v[4];
  float s = 0.0f, ss = 0.0f;
#pragma unroll
  for (int j = 0; j < 4; ++j) {
    v[j] = yr[t + 256 * j];
    s  += v[j];
    ss += v[j] * v[j];
  }
#pragma unroll
  for (int off = 16; off; off >>= 1) {
    s  += __shfl_xor(s,  off, 32);
    ss += __shfl_xor(ss, off, 32);
  }
  __shared__ float sh[16];
  int wid = t >> 5, lane = t & 31;
  if (lane == 0) { sh[wid] = s; sh[8 + wid] = ss; }
  __syncthreads();
  float ts = 0.0f, tss = 0.0f;
#pragma unroll
  for (int w = 0; w < 8; ++w) { ts += sh[w]; tss += sh[8 + w]; }
  float mu  = ts * (1.0f / DI);
  float var = tss * (1.0f / DI) - mu * mu;
  float rs  = rsqrtf(var + 1e-5f);
#pragma unroll
  for (int j = 0; j < 4; ++j) {
    int d = t + 256 * j;
    float z = xz[(size_t)row * (2 * DI) + DI + d];
    float g = z / (1.0f + expf(-z));
    float o = (v[j] - mu) * rs * lnw[d] + lnb[d];
    yln[(size_t)row * DI + d] = (bf16_t)(o * g);
  }
}

// ---------------------------------------------------------------------------
// Host orchestration
// ---------------------------------------------------------------------------
extern "C" void kernel_launch(void* const* d_in, const int* in_sizes, int n_in,
                              void* d_out, int out_size, void* d_ws, size_t ws_size,
                              hipStream_t stream) {
  (void)in_sizes; (void)n_in; (void)out_size; (void)ws_size;
  const float* x      = (const float*)d_in[0];   // (B,L,1024)
  const float* w_in   = (const float*)d_in[1];   // (2048,1024)
  const float* conv_w = (const float*)d_in[2];   // (1024,1,2)
  const float* conv_b = (const float*)d_in[3];   // (1024)
  const float* xproj  = (const float*)d_in[4];   // (66,1024)
  const float* dtw    = (const float*)d_in[5];   // (1024,64)
  const float* dtb    = (const float*)d_in[6];   // (1024)
  const float* Alogs  = (const float*)d_in[7];   // (1024,1)
  const float* Ds     = (const float*)d_in[8];   // (1024)
  const float* lnw    = (const float*)d_in[9];   // (1024)
  const float* lnb    = (const float*)d_in[10];  // (1024)
  const float* w_out  = (const float*)d_in[11];  // (1024,1024)
  float* out = (float*)d_out;                    // (B,L,1024) f32

  char* base = (char*)d_ws;
  size_t off = 0;
  auto alloc = [&](size_t bytes) {
    size_t r = off;
    off += (bytes + 255) & ~(size_t)255;
    return r;
  };
  // 32MB buffer reused: x_bf16 -> xc_conv_bf16 -> y_ln_bf16
  bf16_t* bufA     = (bf16_t*)(base + alloc((size_t)MROWS * DI * 2));
  bf16_t* w_in_bf  = (bf16_t*)(base + alloc((size_t)2 * DI * DM * 2));
  float*  xz       = (float*) (base + alloc((size_t)MROWS * 2 * DI * 4));
  float*  xcc      = (float*) (base + alloc((size_t)MROWS * DI * 4));
  bf16_t* xproj_bf = (bf16_t*)(base + alloc((size_t)96 * DI * 2));
  float*  xdbl     = (float*) (base + alloc((size_t)MROWS * 96 * 4));
  bf16_t* dts_bf   = (bf16_t*)(base + alloc((size_t)MROWS * 64 * 2));
  bf16_t* dtw_bf   = (bf16_t*)(base + alloc((size_t)DI * 64 * 2));
  float*  dtsfull  = (float*) (base + alloc((size_t)MROWS * DI * 4));
  bf16_t* wout_bf  = (bf16_t*)(base + alloc((size_t)DM * DI * 2));
  float*  chP      = (float*) (base + alloc((size_t)BB * DI * NCHUNK * 4));
  float*  chH      = (float*) (base + alloc((size_t)BB * DI * NCHUNK * 4));
  float*  hin      = (float*) (base + alloc((size_t)BB * DI * NCHUNK * 4));
  float*  ybuf     = (float*) (base + alloc((size_t)MROWS * DI * 4));

  const int TB = 256;
  auto grid1 = [&](int n) { return dim3((unsigned)((n + TB - 1) / TB)); };
  auto gemm_grid = [&](int M, int N, int bn) {
    int waves = (M >> 5) * (N / (16 * bn));
    return dim3((unsigned)((waves + 7) / 8));   // 8 waves / 256-thread block
  };

  // --- converts ---
  f32_to_bf16_k<<<grid1(MROWS * DM), TB, 0, stream>>>(x, bufA, MROWS * DM);
  f32_to_bf16_k<<<grid1(2 * DI * DM), TB, 0, stream>>>(w_in, w_in_bf, 2 * DI * DM);
  f32_to_bf16_k<<<grid1(DI * 64), TB, 0, stream>>>(dtw, dtw_bf, DI * 64);
  f32_to_bf16_k<<<grid1(DM * DI), TB, 0, stream>>>(w_out, wout_bf, DM * DI);
  pad_xproj_k<<<grid1(96 * DI), TB, 0, stream>>>(xproj, xproj_bf, 66, 96, DI);

  // --- G1: xz = x @ w_in.T  (16384 x 2048, K=1024), 32x64 tiles ---
  gemm_bf16_bt<4><<<gemm_grid(MROWS, 2 * DI, 4), TB, 0, stream>>>(
      bufA, w_in_bf, xz, MROWS, 2 * DI, DM);

  // --- conv + SiLU (overwrites bufA with xc_conv bf16) ---
  conv_silu_k<<<grid1(MROWS * DI), TB, 0, stream>>>(xz, conv_w, conv_b, xcc, bufA);

  // --- G2: xdbl = xc_conv @ xproj_pad.T  (16384 x 96, K=1024), 32x32 tiles ---
  gemm_bf16_bt<2><<<gemm_grid(MROWS, 96, 2), TB, 0, stream>>>(
      bufA, xproj_bf, xdbl, MROWS, 96, DI);

  // --- dt projection: dts_full = dts @ dtw.T  (16384 x 1024, K=64) ---
  slice_bf16_k<<<grid1(MROWS * 64), TB, 0, stream>>>(xdbl, dts_bf, MROWS, 96, 64);
  gemm_bf16_bt<4><<<gemm_grid(MROWS, DI, 4), TB, 0, stream>>>(
      dts_bf, dtw_bf, dtsfull, MROWS, DI, 64);

  // --- chunked selective scan ---
  scan_pass1_k<<<grid1(BB * NCHUNK * DI), TB, 0, stream>>>(
      dtsfull, xdbl, xcc, dtb, Alogs, chP, chH);
  scan_pass2_k<<<grid1(BB * DI), TB, 0, stream>>>(chP, chH, hin);
  scan_pass3_k<<<grid1(BB * NCHUNK * DI), TB, 0, stream>>>(
      dtsfull, xdbl, xcc, dtb, Alogs, Ds, hin, ybuf);

  // --- LayerNorm + silu(z) gate (overwrites bufA with yln bf16) ---
  ln_gate_k<<<dim3(MROWS), TB, 0, stream>>>(ybuf, xz, lnw, lnb, bufA);

  // --- G4: out = yln @ w_out.T  (16384 x 1024, K=1024), 32x64 tiles ---
  gemm_bf16_bt<4><<<gemm_grid(MROWS, DM, 4), TB, 0, stream>>>(
      bufA, wout_bf, out, MROWS, DM, DI);
}
